// SimVQ_31086973289213
// MI455X (gfx1250) — matrix operand order repaired and checked by reading, not compiled, not run on hardware
//
#include <hip/hip_runtime.h>

typedef __bf16 bf16_t;
typedef __attribute__((ext_vector_type(16))) __bf16 v16bf;
typedef __attribute__((ext_vector_type(8)))  __bf16 v8bf;
typedef __attribute__((ext_vector_type(8)))  float  v8f;

#define NUM_EMB 16384   // K codes
#define N_ROWS  16384   // 8 * 2048
#define DIM     256
#define KSPLIT  2
#define KHALF   (NUM_EMB / KSPLIT)

// ---------------------------------------------------------------------------
// 1) z (f32) -> bf16 for the WMMA distance GEMM
// ---------------------------------------------------------------------------
__global__ void simvq_zcast_kernel(const float* __restrict__ z,
                                   bf16_t* __restrict__ zb, int n) {
  int i = blockIdx.x * blockDim.x + threadIdx.x;
  if (i < n) zb[i] = (bf16_t)z[i];
}

// ---------------------------------------------------------------------------
// 2) quant_codebook = codebook @ proj_w.T + b   (f32 for exact output),
//    fused bf16 copy + row norm ||c||^2.
// ---------------------------------------------------------------------------
__global__ __launch_bounds__(256)
void simvq_project_kernel(const float* __restrict__ cb,
                          const float* __restrict__ w,     // [D, D], row t = proj_w[t]
                          const float* __restrict__ bias,  // [D]
                          float*  __restrict__ qc,         // [K, D] f32
                          bf16_t* __restrict__ qcb,        // [K, D] bf16
                          float*  __restrict__ cnorm) {    // [K]
  __shared__ float srow[DIM];
  __shared__ float red[256];
  const int r = blockIdx.x;
  const int t = threadIdx.x;
  srow[t] = cb[(size_t)r * DIM + t];
  __syncthreads();
  const float* wt = w + (size_t)t * DIM;
  float acc = bias[t];
#pragma unroll 8
  for (int d = 0; d < DIM; ++d) acc = fmaf(srow[d], wt[d], acc);
  qc[(size_t)r * DIM + t]  = acc;
  qcb[(size_t)r * DIM + t] = (bf16_t)acc;
  red[t] = acc * acc;
  __syncthreads();
  for (int s = 128; s > 0; s >>= 1) {
    if (t < s) red[t] += red[t + s];
    __syncthreads();
  }
  if (t == 0) cnorm[r] = red[0];
}

// ---------------------------------------------------------------------------
// 3) Fused distance GEMM + argmin, software-pipelined.
//    score(n,k) = ||c_k||^2 - 2 * z_n . c_k   (||z||^2 dropped: row-constant)
//    Wave owns 32 z-rows (two 16-row A tiles sharing each B tile -> 2x
//    arithmetic intensity). Grid is (row-blocks, KSPLIT): each block scans
//    half the codebook; partial (min,argmin) merged by a tiny kernel.
//    B register double-buffered so the 16 b128 loads overlap the 16 WMMAs.
// ---------------------------------------------------------------------------
__global__ __launch_bounds__(256)
void simvq_argmin_kernel(const bf16_t* __restrict__ zb,   // [N, D]
                         const bf16_t* __restrict__ qcb,  // [K, D]
                         const float*  __restrict__ cnorm,// [K]
                         float* __restrict__ pbv,         // [KSPLIT, N]
                         int*   __restrict__ pbi) {       // [KSPLIT, N]
  const int lane  = threadIdx.x & 31;
  const int wave  = threadIdx.x >> 5;
  const int row0  = (blockIdx.x * 8 + wave) * 32;
  const int kbase = blockIdx.y * KHALF;
  const int kend  = kbase + KHALF;
  const int half  = lane >> 4;    // 0 or 1
  const int r     = lane & 15;

  // --- Preload A (32 z-rows x 256) as two WMMA 16-bit A tiles:
  // lanes 0-15: row M=r, K chunks {0..7} and {16..23}; lanes 16-31 complement.
  v16bf A0[8], A1[8];
  {
    const bf16_t* zr0 = zb + (size_t)(row0 + r) * DIM;
    const bf16_t* zr1 = zb + (size_t)(row0 + 16 + r) * DIM;
#pragma unroll
    for (int d = 0; d < 8; ++d) {
      const int kb = d * 32 + half * 8;
      v8bf lo0 = *(const v8bf*)(zr0 + kb);
      v8bf hi0 = *(const v8bf*)(zr0 + kb + 16);
      A0[d] = __builtin_shufflevector(lo0, hi0, 0,1,2,3,4,5,6,7,8,9,10,11,12,13,14,15);
      v8bf lo1 = *(const v8bf*)(zr1 + kb);
      v8bf hi1 = *(const v8bf*)(zr1 + kb + 16);
      A1[d] = __builtin_shufflevector(lo1, hi1, 0,1,2,3,4,5,6,7,8,9,10,11,12,13,14,15);
    }
  }

  float bv0[8], bv1[8];
  int   bi0[8], bi1[8];
#pragma unroll
  for (int i = 0; i < 8; ++i) {
    bv0[i] = 3.4e38f; bi0[i] = 0;
    bv1[i] = 3.4e38f; bi1[i] = 0;
  }

  // 16-bit B 32x16 layout: lane = column n -> code row (kt + r); each
  // lane-half holds K chunks {8h..8h+7} and {16+8h..23+8h} -> two b128 loads per 32-chunk.
  auto loadB = [&](int kt, v16bf* B) {
    const bf16_t* crow = qcb + (size_t)(kt + r) * DIM;
#pragma unroll
    for (int d = 0; d < 8; ++d) {
      const int kb = d * 32 + half * 8;
      v8bf lo = *(const v8bf*)(crow + kb);
      v8bf hi = *(const v8bf*)(crow + kb + 16);
      B[d] = __builtin_shufflevector(lo, hi, 0,1,2,3,4,5,6,7,8,9,10,11,12,13,14,15);
    }
  };

  // Four independent accumulator chains (2 per m-tile) -> matrix-pipe ILP.
  auto computeTile = [&](int kt, const v16bf* B) {
    const float cn = cnorm[kt + r];
    v8f c0a = {}, c0b = {}, c1a = {}, c1b = {};
#pragma unroll
    for (int d = 0; d < 4; ++d) {
      c0a = __builtin_amdgcn_wmma_f32_16x16x32_bf16(false, A0[d],     false, B[d],
                                                    (short)0, c0a, false, false);
      c1a = __builtin_amdgcn_wmma_f32_16x16x32_bf16(false, A1[d],     false, B[d],
                                                    (short)0, c1a, false, false);
    }
#pragma unroll
    for (int d = 0; d < 4; ++d) {
      c0b = __builtin_amdgcn_wmma_f32_16x16x32_bf16(false, A0[d + 4], false, B[d + 4],
                                                    (short)0, c0b, false, false);
      c1b = __builtin_amdgcn_wmma_f32_16x16x32_bf16(false, A1[d + 4], false, B[d + 4],
                                                    (short)0, c1b, false, false);
    }
    // C layout: slot i -> row m = i + 8*half, column n = r (fixed per lane).
#pragma unroll
    for (int i = 0; i < 8; ++i) {
      const float s0 = fmaf(-2.0f, c0a[i] + c0b[i], cn);
      if (s0 < bv0[i]) { bv0[i] = s0; bi0[i] = kt + r; }
      const float s1 = fmaf(-2.0f, c1a[i] + c1b[i], cn);
      if (s1 < bv1[i]) { bv1[i] = s1; bi1[i] = kt + r; }
    }
  };

  v16bf B0[8], B1[8];
  loadB(kbase, B0);
  for (int kt = kbase; kt < kend; kt += 32) {
    loadB(kt + 16, B1);                        // overlap with compute on B0
    computeTile(kt, B0);
    int nk = kt + 32; nk = (nk == kend) ? kbase : nk;  // uniform wrap for tail
    loadB(nk, B0);
    computeTile(kt + 16, B1);
  }

  // Min+argmin across the 16 lanes sharing each row m (xor within halves).
  float* pv = pbv + (size_t)blockIdx.y * N_ROWS;
  int*   pi = pbi + (size_t)blockIdx.y * N_ROWS;
#pragma unroll
  for (int i = 0; i < 8; ++i) {
    float v0 = bv0[i]; int ix0 = bi0[i];
    float v1 = bv1[i]; int ix1 = bi1[i];
#pragma unroll
    for (int off = 1; off < 16; off <<= 1) {
      const float ov0 = __shfl_xor(v0, off, 32);
      const int   oi0 = __shfl_xor(ix0, off, 32);
      if (ov0 < v0 || (ov0 == v0 && oi0 < ix0)) { v0 = ov0; ix0 = oi0; }
      const float ov1 = __shfl_xor(v1, off, 32);
      const int   oi1 = __shfl_xor(ix1, off, 32);
      if (ov1 < v1 || (ov1 == v1 && oi1 < ix1)) { v1 = ov1; ix1 = oi1; }
    }
    if (r == 0) {
      const int m = i + half * 8;
      pv[row0 + m]      = v0;  pi[row0 + m]      = ix0;
      pv[row0 + 16 + m] = v1;  pi[row0 + 16 + m] = ix1;
    }
  }
}

// ---------------------------------------------------------------------------
// 4) merge the KSPLIT partial argmins (half-0 indices < half-1 -> ties pick 0)
// ---------------------------------------------------------------------------
__global__ void simvq_merge_kernel(const float* __restrict__ pbv,
                                   const int*   __restrict__ pbi,
                                   int* __restrict__ idx_out) {
  const int n = blockIdx.x * blockDim.x + threadIdx.x;
  if (n >= N_ROWS) return;
  const float v0 = pbv[n];
  const float v1 = pbv[N_ROWS + n];
  const int   i0 = pbi[n];
  const int   i1 = pbi[N_ROWS + n];
  idx_out[n] = (v1 < v0) ? i1 : i0;
}

// ---------------------------------------------------------------------------
// 5) zero the loss accumulator
// ---------------------------------------------------------------------------
__global__ void simvq_loss_init_kernel(float* loss) { *loss = 0.0f; }

// ---------------------------------------------------------------------------
// 6) gather quantized rows (exact f32 codebook) + vq_loss accumulation.
//    vq_loss = (1 + 0.25) * mean((q - z)^2)
// ---------------------------------------------------------------------------
__global__ __launch_bounds__(256)
void simvq_gather_loss_kernel(const float* __restrict__ z,
                              const float* __restrict__ qc,
                              const int*   __restrict__ idx,
                              float* __restrict__ qout,
                              float* __restrict__ loss) {
  __shared__ float red[256];
  const int n = blockIdx.x;
  const int t = threadIdx.x;
  const int k = idx[n];
  const float q  = qc[(size_t)k * DIM + t];
  const float zv = z[(size_t)n * DIM + t];
  qout[(size_t)n * DIM + t] = q;
  const float d = q - zv;
  red[t] = d * d;
  __syncthreads();
  for (int s = 128; s > 0; s >>= 1) {
    if (t < s) red[t] += red[t + s];
    __syncthreads();
  }
  if (t == 0)
    atomicAdd(loss, red[0] * (1.25f / (float)((size_t)N_ROWS * DIM)));
}

// ---------------------------------------------------------------------------
extern "C" void kernel_launch(void* const* d_in, const int* in_sizes, int n_in,
                              void* d_out, int out_size, void* d_ws, size_t ws_size,
                              hipStream_t stream) {
  const float* z    = (const float*)d_in[0];  // [8,2048,256]
  const float* cb   = (const float*)d_in[1];  // [16384,256]
  const float* w    = (const float*)d_in[2];  // [256,256]
  const float* bias = (const float*)d_in[3];  // [256]

  // workspace layout (~32.4 MB)
  float*  qc_f32 = (float*)d_ws;                                   // 16 MB
  bf16_t* qc_bf  = (bf16_t*)(qc_f32 + (size_t)NUM_EMB * DIM);      //  8 MB
  bf16_t* z_bf   = qc_bf + (size_t)NUM_EMB * DIM;                  //  8 MB
  float*  cnorm  = (float*)(z_bf + (size_t)N_ROWS * DIM);          // 64 KB
  float*  pbv    = cnorm + NUM_EMB;                                // 128 KB
  int*    pbi    = (int*)(pbv + (size_t)KSPLIT * N_ROWS);          // 128 KB

  // output layout: [quantized 4194304 f32][vq_loss 1 f32][indices 16384 i32]
  float* outq = (float*)d_out;
  float* loss = outq + (size_t)N_ROWS * DIM;
  int*   idx  = (int*)(loss + 1);

  simvq_zcast_kernel<<<(N_ROWS * DIM) / 256, 256, 0, stream>>>(z, z_bf, N_ROWS * DIM);
  simvq_project_kernel<<<NUM_EMB, 256, 0, stream>>>(cb, w, bias, qc_f32, qc_bf, cnorm);
  simvq_argmin_kernel<<<dim3(N_ROWS / 256, KSPLIT), 256, 0, stream>>>(z_bf, qc_bf, cnorm,
                                                                     pbv, pbi);
  simvq_merge_kernel<<<N_ROWS / 256, 256, 0, stream>>>(pbv, pbi, idx);
  simvq_loss_init_kernel<<<1, 1, 0, stream>>>(loss);
  simvq_gather_loss_kernel<<<N_ROWS, 256, 0, stream>>>(z, qc_f32, idx, outq, loss);
}
